// AGREE_12773232738622
// MI455X (gfx1250) — matrix operand order repaired and checked by reading, not compiled, not run on hardware
//
#include <hip/hip_runtime.h>
#include <hip/hip_bf16.h>

typedef __attribute__((ext_vector_type(16))) _Float16 v16h;
typedef __attribute__((ext_vector_type(8)))  _Float16 v8h;
typedef __attribute__((ext_vector_type(2)))  _Float16 h2v;
typedef __attribute__((ext_vector_type(8)))  float    v8f;

#define MAXM   50
#define MPAD   64
#define EMB    64
#define ATT_H  16
#define PRED_H 8
#define NEG_BIG (-1.0e30f)

// ---------------- lane-reduction helpers (VALU permlane, no LDS) ----------------
#if __has_builtin(__builtin_amdgcn_permlane16) && __has_builtin(__builtin_amdgcn_permlanex16)
#define USE_PERMLANE 1
#endif

#ifdef USE_PERMLANE
__device__ __forceinline__ float pl16(float x, unsigned lo, unsigned hi) {
    unsigned u = __builtin_bit_cast(unsigned, x);
    unsigned r = __builtin_amdgcn_permlane16(u, u, lo, hi, false, false);
    return __builtin_bit_cast(float, r);
}
__device__ __forceinline__ float plx16(float x) {   // swap 16-lane halves (xor 16)
    unsigned u = __builtin_bit_cast(unsigned, x);
    unsigned r = __builtin_amdgcn_permlanex16(u, u, 0x76543210u, 0xFEDCBA98u, false, false);
    return __builtin_bit_cast(float, r);
}
// sum across each 16-lane half (xor-butterfly 1,2,4,8)
__device__ __forceinline__ float half16_sum(float v) {
    v += pl16(v, 0x67452301u, 0xEFCDAB89u);   // xor 1
    v += pl16(v, 0x54761032u, 0xDCFE98BAu);   // xor 2
    v += pl16(v, 0x32107654u, 0xBA98FEDCu);   // xor 4
    v += pl16(v, 0xFEDCBA98u, 0x76543210u);   // xor 8
    return v;
}
__device__ __forceinline__ float wave_sum(float v) { v = half16_sum(v); return v + plx16(v); }
__device__ __forceinline__ float wave_max(float v) {
    v = fmaxf(v, pl16(v, 0x67452301u, 0xEFCDAB89u));
    v = fmaxf(v, pl16(v, 0x54761032u, 0xDCFE98BAu));
    v = fmaxf(v, pl16(v, 0x32107654u, 0xBA98FEDCu));
    v = fmaxf(v, pl16(v, 0xFEDCBA98u, 0x76543210u));
    return fmaxf(v, plx16(v));
}
#else
__device__ __forceinline__ float half16_sum(float v) {
    v += __shfl_xor(v, 1, 32); v += __shfl_xor(v, 2, 32);
    v += __shfl_xor(v, 4, 32); v += __shfl_xor(v, 8, 32);
    return v;
}
__device__ __forceinline__ float wave_sum(float v) { v = half16_sum(v); return v + __shfl_xor(v, 16, 32); }
__device__ __forceinline__ float wave_max(float v) {
    for (int off = 1; off < 32; off <<= 1) v = fmaxf(v, __shfl_xor(v, off, 32));
    return v;
}
#endif

// One wave32 per batch group.
// Attention GEMM: item half of cat@att_w1 is constant over members -> folded into a
// per-group f32 bias; WMMA runs only over the member half (K=64 -> 2 chunks of
// v_wmma_f32_16x16x32_f16 per 16-row tile, 4 tiles for 50->64 padded members).
__launch_bounds__(32)
__global__ void agree_fused_kernel(
    const int*           __restrict__ member_idx,   // [B, MAXM]
    const unsigned char* __restrict__ member_mask,  // [B, MAXM] (bool)
    const int*           __restrict__ item_inputs,  // [B]
    const float*         __restrict__ user_table,   // [NU, EMB]
    const float*         __restrict__ item_table,   // [NI, EMB]
    const float*         __restrict__ att_w1,       // [2*EMB, ATT_H]
    const float*         __restrict__ att_b1,       // [ATT_H]
    const float*         __restrict__ att_w2,       // [ATT_H, 1]
    const float*         __restrict__ att_b2,       // [1]
    const float*         __restrict__ pred_w1,      // [3*EMB, PRED_H]
    const float*         __restrict__ pred_b1,      // [PRED_H]
    const float*         __restrict__ pred_w2,      // [PRED_H, 1]
    const float*         __restrict__ pred_b2,      // [1]
    float*               __restrict__ out)          // [B, 1]
{
    __shared__ __align__(16) _Float16 catH[MPAD][EMB];  // 8 KB: f16 members, padded rows
    __shared__ float    memF[MAXM][EMB];                // 12.8 KB: f32 members (pooling)
    __shared__ float    itemF[EMB];
    __shared__ float    logitsS[MPAD];
    __shared__ float    wtsS[MAXM];
    __shared__ int      midx[MAXM];

    const int b    = blockIdx.x;
    const int lane = threadIdx.x;            // 0..31, wave32
    const int hsel = lane & 15;              // N / hidden index
    const int hi16 = lane >> 4;              // which 16-lane half

    // ---- stage indices + item embedding (float2 per lane) ----
    if (lane < MAXM)      midx[lane]      = member_idx[b * MAXM + lane];
    if (lane + 32 < MAXM) midx[lane + 32] = member_idx[b * MAXM + lane + 32];
    const int it = item_inputs[b];
    const float2 itv = ((const float2*)(item_table + (long)it * EMB))[lane];
    ((float2*)itemF)[lane] = itv;
    __syncthreads();

    // ---- gather members: one b64 global load / lane / member ----
    for (int m = 0; m < MAXM; ++m) {
        const long u = (long)midx[m];
        const float2 f = ((const float2*)(user_table + u * EMB))[lane];
        ((float2*)&memF[m][0])[lane] = f;
        h2v h; h.x = (_Float16)f.x; h.y = (_Float16)f.y;
        ((h2v*)&catH[m][0])[lane] = h;
    }
    #pragma unroll
    for (int m = MAXM; m < MPAD; ++m) {
        h2v z; z.x = (_Float16)0.f; z.y = (_Float16)0.f;
        ((h2v*)&catH[m][0])[lane] = z;
    }
    __syncthreads();

    // ---- per-group f32 bias: item @ att_w1[64:128,:] + att_b1 (exact, once) ----
    // lanes 0-15: h=lane, k=0..31 ; lanes 16-31: h=lane-16, k=32..63 ; then swap-add.
    float itacc = 0.f;
    {
        const int kbase = hi16 * 32;
        #pragma unroll
        for (int kk = 0; kk < 32; ++kk) {
            const int k = kbase + kk;
            itacc = fmaf(itemF[k], att_w1[(EMB + k) * ATT_H + hsel], itacc);
        }
    }
#ifdef USE_PERMLANE
    itacc += plx16(itacc);
#else
    itacc += __shfl_xor(itacc, 16, 32);
#endif
    const float hbias = itacc + att_b1[hsel];   // valid in all lanes for h = hsel
    const float w2l   = att_w2[hsel];
    const float b2s   = att_b2[0];

    // ---- B fragments for att_w1[0:64,:] (2 K-chunks, shared by all tiles) ----
    // 16-bit B KxN layout: lanes 0-15 hold K = 32c + {0..7,16..23}, lanes 16-31 K = 32c + {8..15,24..31}.
    v16h bf[2];
    {
        const int kh = hi16 * 8;
        #pragma unroll
        for (int c = 0; c < 2; ++c) {
            #pragma unroll
            for (int j = 0; j < 16; ++j) {
                const int K = 32 * c + kh + (j & 7) + ((j >> 3) << 4);
                bf[c][j] = (_Float16)att_w1[K * ATT_H + hsel];
            }
        }
    }

    // ---- attention GEMM: 4 M-tiles x 2 K-chunks ----
    #pragma unroll
    for (int t = 0; t < 4; ++t) {
        v8f d = {};
        const int M = hsel + 16 * t;
        const v8h* row = (const v8h*)&catH[M][0];    // 16-byte units along the row
        #pragma unroll
        for (int c = 0; c < 2; ++c) {
            // A layout per lane: two contiguous 8-half runs at kbase and kbase+16
            const int kbase = 32 * c + 8 * hi16;
            const v8h lo = row[(kbase >> 3)];
            const v8h hh = row[(kbase >> 3) + 2];
            const v16h a = __builtin_shufflevector(lo, hh,
                0, 1, 2, 3, 4, 5, 6, 7, 8, 9, 10, 11, 12, 13, 14, 15);
            d = __builtin_amdgcn_wmma_f32_16x16x32_f16(
                    false, a, false, bf[c], (short)0, d, false, false);
        }
        // D: vgpr r -> row (r + 8*hi16 + 16t), col h = hsel.
        #pragma unroll
        for (int r = 0; r < 8; ++r) {
            float v = fmaxf(d[r] + hbias, 0.f) * w2l;
            v = half16_sum(v);
            if (hsel == 0)
                logitsS[16 * t + 8 * hi16 + r] = v + b2s;
        }
    }
    __syncthreads();

    // ---- masked softmax over MAXM members (2 entries per lane) ----
    const int m0 = lane, m1 = lane + 32;
    const bool v0 = (m0 < MAXM) && (member_mask[b * MAXM + m0] != 0);
    const bool v1 = (m1 < MAXM) && (member_mask[b * MAXM + m1] != 0);
    const float l0 = v0 ? logitsS[m0] : NEG_BIG;
    const float l1 = v1 ? logitsS[m1] : NEG_BIG;
    const float mx = wave_max(fmaxf(l0, l1));
    const float e0 = v0 ? __expf(l0 - mx) : 0.f;
    const float e1 = v1 ? __expf(l1 - mx) : 0.f;
    const float inv = 1.f / wave_sum(e0 + e1);
    if (m0 < MAXM) wtsS[m0] = e0 * inv;
    if (m1 < MAXM) wtsS[m1] = e1 * inv;
    __syncthreads();

    // ---- attention-weighted pooling: columns c0=2*lane, c1=2*lane+1 ----
    float g0 = 0.f, g1 = 0.f;
    for (int m = 0; m < MAXM; ++m) {
        const float w = wtsS[m];
        const float2 f = ((const float2*)&memF[m][0])[lane];
        g0 = fmaf(w, f.x, g0);
        g1 = fmaf(w, f.y, g1);
    }

    // ---- prediction MLP: new = [g*item, g, item] (192) @ pred_w1 (192x8) ----
    const int c0 = 2 * lane, c1 = 2 * lane + 1;
    const float it0 = itv.x, it1 = itv.y;
    float p[PRED_H];
    #pragma unroll
    for (int j = 0; j < PRED_H; ++j) {
        p[j] = fmaf(g0 * it0, pred_w1[(      c0) * PRED_H + j],
               fmaf(g0,       pred_w1[( 64 + c0) * PRED_H + j],
               fmaf(it0,      pred_w1[(128 + c0) * PRED_H + j],
               fmaf(g1 * it1, pred_w1[(      c1) * PRED_H + j],
               fmaf(g1,       pred_w1[( 64 + c1) * PRED_H + j],
                    it1 *     pred_w1[(128 + c1) * PRED_H + j])))));
        p[j] = wave_sum(p[j]);
    }
    if (lane == 0) {
        float acc = pred_b2[0];
        #pragma unroll
        for (int j = 0; j < PRED_H; ++j) {
            const float ph = fmaxf(p[j] + pred_b1[j], 0.f);
            acc = fmaf(ph, pred_w2[j], acc);
        }
        out[b] = 1.f / (1.f + __expf(-acc));
    }
}

extern "C" void kernel_launch(void* const* d_in, const int* in_sizes, int n_in,
                              void* d_out, int out_size, void* d_ws, size_t ws_size,
                              hipStream_t stream) {
    const int*           member_idx  = (const int*)d_in[0];
    const unsigned char* member_mask = (const unsigned char*)d_in[1];  // jax bool = 1 byte
    const int*           item_inputs = (const int*)d_in[2];
    const float*         user_table  = (const float*)d_in[3];
    const float*         item_table  = (const float*)d_in[4];
    const float*         att_w1      = (const float*)d_in[5];
    const float*         att_b1      = (const float*)d_in[6];
    const float*         att_w2      = (const float*)d_in[7];
    const float*         att_b2      = (const float*)d_in[8];
    const float*         pred_w1     = (const float*)d_in[9];
    const float*         pred_b1     = (const float*)d_in[10];
    const float*         pred_w2     = (const float*)d_in[11];
    const float*         pred_b2     = (const float*)d_in[12];
    float*               out         = (float*)d_out;

    const int B = in_sizes[2];   // item_inputs is [B]
    agree_fused_kernel<<<B, 32, 0, stream>>>(
        member_idx, member_mask, item_inputs, user_table, item_table,
        att_w1, att_b1, att_w2, att_b2, pred_w1, pred_b1, pred_w2, pred_b2, out);
}
